// TinyMamba3_45638322487935
// MI455X (gfx1250) — matrix operand order repaired and checked by reading, not compiled, not run on hardware
//
#include <hip/hip_runtime.h>
#include <hip/hip_bf16.h>

// ---------------- problem constants (from reference) ----------------
#define BATCH    16
#define SEQ      2048
#define IN_DIM   57
#define D_MODEL  64
#define HEADDIM  32
#define D_STATE  32
#define NHEADS   4
#define D_INNER  128
#define D_CONV   4
#define D_XBC    192           // D_INNER + 2*D_STATE
#define D_IN_PROJ 324          // 2*D_INNER + 2*D_STATE + NHEADS
#define D_PROJ_PAD 384         // zxbcdt row stride padded to multiple of 64
#define OUT_DIM  6
#define NTOK     (BATCH*SEQ)   // 32768

typedef __attribute__((ext_vector_type(16))) __bf16 v16bf;
typedef __attribute__((ext_vector_type(8)))  float  v8f;

__device__ __forceinline__ float sigmoidf_(float x) { return 1.0f / (1.0f + expf(-x)); }

__device__ __forceinline__ unsigned int pack_bf16(float a, float b) {
    union { __bf16 h[2]; unsigned int u; } p;
    p.h[0] = (__bf16)a; p.h[1] = (__bf16)b;      // -> v_cvt_pk_bf16_f32
    return p.u;
}

// =====================================================================
// Shape-exact WMMA GEMM, no LDS, no barriers, no bounds checks.
//   C[M, ldc] = A[M, K](f32) @ BT[N, K](bf16, pre-transposed)  (+bias)
// Requirements: M % 64 == 0 (grid.y = M/64), N % 64 == 0 (grid.x = N/64),
// K % 32 == 0 and known at compile time (fully unrolled K loop).
// Block = 256 threads = 8 waves; wave -> one 16-row A fragment (mi=wv>>1)
// and two 16-col B fragments (ni = (wv&1)*2 {,+1}); 2 WMMAs per K-step.
// Fragment packing: A 16x32 in contiguous k = 16*half + s order (not the ISA table order);
// B in the same k order; C/D per the 8-VGPR table.
// =====================================================================
template<int K, bool BIAS>
__global__ __launch_bounds__(256)
void gemm_ws_wmma(const float* __restrict__ A,
                  const __bf16* __restrict__ BT,
                  const float* __restrict__ bias,
                  float* __restrict__ C, int ldc)
{
    const int lane = threadIdx.x & 31;
    const int wv   = threadIdx.x >> 5;
    const int m0   = blockIdx.y * 64;
    const int n0   = blockIdx.x * 64;
    const int mi   = wv >> 1;
    const int ni0  = (wv & 1) * 2;
    const int ni1  = ni0 + 1;
    const int fr   = lane & 15;
    const int half = lane >> 4;

    const float* arow = A + (size_t)(m0 + mi * 16 + fr) * K;
    const unsigned int* b0 = (const unsigned int*)(BT + (size_t)(n0 + ni0 * 16 + fr) * K);
    const unsigned int* b1 = (const unsigned int*)(BT + (size_t)(n0 + ni1 * 16 + fr) * K);

    v8f acc0 = {}; v8f acc1 = {};

    #pragma unroll
    for (int k0 = 0; k0 < K; k0 += 32) {
        // ---- A fragment: lanes 0-15 hold K 0..15, lanes 16-31 hold K 16..31.
        // 16 consecutive floats -> 4 aligned float4 loads.
        float4 a0 = *(const float4*)(arow + k0 + half * 16);
        float4 a1 = *(const float4*)(arow + k0 + half * 16 + 4);
        float4 a2 = *(const float4*)(arow + k0 + half * 16 + 8);
        float4 a3 = *(const float4*)(arow + k0 + half * 16 + 12);
        union { v16bf v; unsigned int u[8]; } af;
        af.u[0] = pack_bf16(a0.x, a0.y); af.u[1] = pack_bf16(a0.z, a0.w);
        af.u[2] = pack_bf16(a1.x, a1.y); af.u[3] = pack_bf16(a1.z, a1.w);
        af.u[4] = pack_bf16(a2.x, a2.y); af.u[5] = pack_bf16(a2.z, a2.w);
        af.u[6] = pack_bf16(a3.x, a3.y); af.u[7] = pack_bf16(a3.z, a3.w);

        // ---- B fragments: lane half selects K half; 8 consecutive dwords
        // of bf16 pairs -> 2 aligned uint4 loads per fragment.
        const int bq = (k0 + half * 16) >> 1;     // dword index
        uint4 q0 = *(const uint4*)(b0 + bq);
        uint4 q1 = *(const uint4*)(b0 + bq + 4);
        uint4 r0 = *(const uint4*)(b1 + bq);
        uint4 r1 = *(const uint4*)(b1 + bq + 4);
        union { v16bf v; unsigned int u[8]; } bf0, bf1;
        bf0.u[0] = q0.x; bf0.u[1] = q0.y; bf0.u[2] = q0.z; bf0.u[3] = q0.w;
        bf0.u[4] = q1.x; bf0.u[5] = q1.y; bf0.u[6] = q1.z; bf0.u[7] = q1.w;
        bf1.u[0] = r0.x; bf1.u[1] = r0.y; bf1.u[2] = r0.z; bf1.u[3] = r0.w;
        bf1.u[4] = r1.x; bf1.u[5] = r1.y; bf1.u[6] = r1.z; bf1.u[7] = r1.w;

        acc0 = __builtin_amdgcn_wmma_f32_16x16x32_bf16(false, af.v, false, bf0.v,
                                                       (short)0, acc0, false, false);
        acc1 = __builtin_amdgcn_wmma_f32_16x16x32_bf16(false, af.v, false, bf1.v,
                                                       (short)0, acc1, false, false);
    }

    float bias0 = 0.0f, bias1 = 0.0f;
    if (BIAS) {
        bias0 = bias[n0 + ni0 * 16 + fr];
        bias1 = bias[n0 + ni1 * 16 + fr];
    }
    // C/D layout: VGPR r -> (M = r + half*8, N = fr)
    #pragma unroll
    for (int r = 0; r < 8; ++r) {
        int m = m0 + mi * 16 + r + half * 8;
        C[(size_t)m * ldc + (n0 + ni0 * 16 + fr)] = acc0[r] + bias0;
        C[(size_t)m * ldc + (n0 + ni1 * 16 + fr)] = acc1[r] + bias1;
    }
}

// =====================================================================
// Prep: pad x [NTOK,57] -> xpad [NTOK,64] (zero-filled tail).
// =====================================================================
__global__ void pad_x_kernel(const float* __restrict__ x, float* __restrict__ xpad)
{
    int idx = blockIdx.x * 256 + threadIdx.x;
    if (idx >= NTOK * 64) return;
    int k = idx & 63, t = idx >> 6;
    xpad[idx] = (k < IN_DIM) ? x[(size_t)t * IN_DIM + k] : 0.0f;
}

// =====================================================================
// Prep: W [K,N] f32 -> WT [Npad, Kpad] bf16, transposed + zero-padded.
// =====================================================================
__global__ void transpose_pad_bf16(const float* __restrict__ W, __bf16* __restrict__ WT,
                                   int K, int N, int Kpad, int Npad)
{
    int idx = blockIdx.x * 256 + threadIdx.x;
    if (idx >= Kpad * Npad) return;
    int k = idx % Kpad, n = idx / Kpad;
    float v = (k < K && n < N) ? W[(size_t)k * N + n] : 0.0f;
    WT[idx] = (__bf16)v;
}

// =====================================================================
// Causal depthwise conv (k=4) + bias + SiLU on the xBC slice of zxbcdt.
// =====================================================================
__global__ void conv_silu_kernel(const float* __restrict__ zxbcdt,
                                 const float* __restrict__ conv_w,
                                 const float* __restrict__ conv_b,
                                 float* __restrict__ xbc_out)
{
    int idx = blockIdx.x * blockDim.x + threadIdx.x;
    if (idx >= NTOK * D_XBC) return;
    int c = idx % D_XBC;
    int t = (idx / D_XBC) % SEQ;
    int b = idx / (D_XBC * SEQ);
    const float* base = zxbcdt + (size_t)(b * SEQ) * D_PROJ_PAD + D_INNER + c;
    float acc = conv_b[c];
    #pragma unroll
    for (int j = 0; j < D_CONV; ++j) {
        int tt = t - (D_CONV - 1) + j;
        float xv = (tt >= 0) ? base[(size_t)tt * D_PROJ_PAD] : 0.0f;
        acc = fmaf(conv_w[c * D_CONV + j], xv, acc);
    }
    xbc_out[idx] = acc * sigmoidf_(acc);   // silu
}

// =====================================================================
// Precompute dt = softplus(dt_raw + dt_bias) and decay = exp(dt * A).
// =====================================================================
__global__ void dt_kernel(const float* __restrict__ zxbcdt,
                          const float* __restrict__ dt_bias,
                          const float* __restrict__ A_log,
                          float* __restrict__ dtv, float* __restrict__ decay)
{
    int idx = blockIdx.x * blockDim.x + threadIdx.x;
    if (idx >= BATCH * NHEADS * SEQ) return;
    int t = idx % SEQ;
    int h = (idx / SEQ) % NHEADS;
    int b = idx / (SEQ * NHEADS);
    float raw = zxbcdt[(size_t)(b * SEQ + t) * D_PROJ_PAD + D_INNER + D_XBC + h] + dt_bias[h];
    float dt = (raw > 20.0f) ? raw : log1pf(expf(raw));
    float A = -expf(A_log[h]);
    dtv[idx]   = dt;           // layout [b][h][t]
    decay[idx] = expf(dt * A);
}

// =====================================================================
// Recurrent scan. decay is scalar per (b,h,t) -> rank-1 state update.
// Block = one (b,h); wave = headdim p; lane = state n; 1 state float/lane.
// Software-pipelined loads + clamped global_prefetch hide L2 latency in
// the serial t loop. Writes yraw = scan_y + D[h]*x.
// =====================================================================
__global__ __launch_bounds__(1024)
void scan_kernel(const float* __restrict__ xbc, const float* __restrict__ dtv,
                 const float* __restrict__ decay, const float* __restrict__ Dp,
                 float* __restrict__ yraw)
{
    const int bh = blockIdx.x;          // 0..63
    const int b  = bh >> 2;
    const int hh = bh & 3;
    const int p  = threadIdx.x >> 5;    // wave id = headdim index
    const int n  = threadIdx.x & 31;    // lane    = state index

    const float  Dh  = Dp[hh];
    const float* xb  = xbc   + (size_t)b * SEQ * D_XBC;
    const float* dtp = dtv   + (size_t)(b * NHEADS + hh) * SEQ;
    const float* dcp = decay + (size_t)(b * NHEADS + hh) * SEQ;
    float*       yb  = yraw  + (size_t)b * SEQ * D_INNER + hh * HEADDIM + p;

    float hst = 0.0f;
    float x_c  = xb[hh * HEADDIM + p];
    float B_c  = xb[D_INNER + n];
    float C_c  = xb[D_INNER + D_STATE + n];
    float dt_c = dtp[0];
    float dc_c = dcp[0];

    for (int t = 0; t < SEQ; ++t) {
        int tn = (t + 1 < SEQ) ? (t + 1) : t;       // clamped next (values unused at t=SEQ-1)
        const float* row = xb + (size_t)tn * D_XBC;
        float x_n  = row[hh * HEADDIM + p];
        float B_n  = row[D_INNER + n];
        float C_n  = row[D_INNER + D_STATE + n];
        float dt_n = dtp[tn];
        float dc_n = dcp[tn];

        int tp = (t + 8 < SEQ) ? (t + 8) : (SEQ - 1);
        __builtin_prefetch(xb + (size_t)tp * D_XBC + 4 * n, 0, 1);  // global_prefetch_b8

        float coef = dt_c * x_c;
        hst = fmaf(hst, dc_c, coef * B_c);          // h = h*decay + dt*x[p]*B[n]
        float val = hst * C_c;                      // partial of y[p] = sum_n h*C[n]
        #pragma unroll
        for (int o = 16; o > 0; o >>= 1) val += __shfl_xor(val, o, 32);
        if (n == 0) yb[(size_t)t * D_INNER] = val + Dh * x_c;

        x_c = x_n; B_c = B_n; C_c = C_n; dt_c = dt_n; dc_c = dc_n;
    }
}

// =====================================================================
// Gate with silu(z), RMS-norm over 128 channels, scale by norm_w.
// One wave per token, 4 channels per lane, in-place on yraw.
// =====================================================================
__global__ __launch_bounds__(256)
void gate_norm_kernel(float* __restrict__ yraw, const float* __restrict__ zxbcdt,
                      const float* __restrict__ norm_w)
{
    int token = blockIdx.x * 8 + (threadIdx.x >> 5);
    int lane  = threadIdx.x & 31;
    if (token >= NTOK) return;
    float*       y = yraw   + (size_t)token * D_INNER;
    const float* z = zxbcdt + (size_t)token * D_PROJ_PAD;   // z = first 128 ch
    float g[4]; float ss = 0.0f;
    #pragma unroll
    for (int j = 0; j < 4; ++j) {
        int c = lane + 32 * j;
        float zz = z[c];
        float gv = y[c] * (zz * sigmoidf_(zz));
        g[j] = gv;
        ss = fmaf(gv, gv, ss);
    }
    #pragma unroll
    for (int o = 16; o > 0; o >>= 1) ss += __shfl_xor(ss, o, 32);
    float scale = rsqrtf(ss * (1.0f / D_INNER) + 1e-5f);
    #pragma unroll
    for (int j = 0; j < 4; ++j) {
        int c = lane + 32 * j;
        y[c] = g[j] * scale * norm_w[c];
    }
}

// =====================================================================
// Mean-pool over L then tiny classifier (64x6).
// =====================================================================
__global__ __launch_bounds__(256)
void pool_cls_kernel(const float* __restrict__ yout, const float* __restrict__ W_cls,
                     const float* __restrict__ b_cls, float* __restrict__ out)
{
    __shared__ float red[256];
    __shared__ float pooled[D_MODEL];
    int b   = blockIdx.x;
    int tid = threadIdx.x;
    int d   = tid & 63, ts = tid >> 6;
    float acc = 0.0f;
    const float* yb = yout + (size_t)b * SEQ * D_MODEL;
    for (int t = ts; t < SEQ; t += 4) acc += yb[(size_t)t * D_MODEL + d];
    red[tid] = acc;
    __syncthreads();
    if (ts == 0)
        pooled[d] = (red[d] + red[64 + d] + red[128 + d] + red[192 + d]) * (1.0f / SEQ);
    __syncthreads();
    if (tid < OUT_DIM) {
        float s = b_cls[tid];
        for (int dd = 0; dd < D_MODEL; ++dd)
            s = fmaf(pooled[dd], W_cls[dd * OUT_DIM + tid], s);
        out[b * OUT_DIM + tid] = s;
    }
}

// =====================================================================
extern "C" void kernel_launch(void* const* d_in, const int* in_sizes, int n_in,
                              void* d_out, int out_size, void* d_ws, size_t ws_size,
                              hipStream_t stream)
{
    (void)in_sizes; (void)n_in; (void)out_size; (void)ws_size;
    const float* x       = (const float*)d_in[0];
    const float* W_lin   = (const float*)d_in[1];
    const float* b_lin   = (const float*)d_in[2];
    const float* W_in    = (const float*)d_in[3];
    const float* conv_w  = (const float*)d_in[4];
    const float* conv_b  = (const float*)d_in[5];
    const float* dt_bias = (const float*)d_in[6];
    const float* A_log   = (const float*)d_in[7];
    const float* Dp      = (const float*)d_in[8];
    const float* norm_w  = (const float*)d_in[9];
    const float* W_out   = (const float*)d_in[10];
    const float* W_cls   = (const float*)d_in[11];
    const float* b_cls   = (const float*)d_in[12];
    float* out = (float*)d_out;

    // ---- workspace layout (~110 MB f32 + tiny bf16 weights) ----
    float* ws     = (float*)d_ws;
    size_t o      = 0;
    float* zxbcdt = ws + o; o += (size_t)NTOK * D_PROJ_PAD;       // 12.6M (stride 384)
    float* xbc    = ws + o; o += (size_t)NTOK * D_XBC;            //  6.3M
    float* yraw   = ws + o; o += (size_t)NTOK * D_INNER;          //  4.2M
    float* yout   = ws + o; o += (size_t)NTOK * D_MODEL;          //  2.1M
    float* xpad   = ws + o; o += (size_t)NTOK * 64;               //  2.1M
    float* dtv    = ws + o; o += (size_t)BATCH * NHEADS * SEQ;    //  131K
    float* decay  = ws + o; o += (size_t)BATCH * NHEADS * SEQ;    //  131K
    __bf16* WlinT = (__bf16*)(ws + o);            // [64][64]
    __bf16* WinT  = WlinT + 64 * 64;              // [384][64]
    __bf16* WoutT = WinT + 384 * 64;              // [64][128]
    float* h      = yraw;   // alias: h fully consumed by GEMM2 before scan writes yraw

    dim3 blk(256);

    // ---- prep: pad x, transpose+convert weights to bf16 ----
    pad_x_kernel<<<(NTOK * 64) / 256, blk, 0, stream>>>(x, xpad);
    transpose_pad_bf16<<<(64 * 64 + 255) / 256, blk, 0, stream>>>(W_lin, WlinT, IN_DIM, 64, 64, 64);
    transpose_pad_bf16<<<(384 * 64 + 255) / 256, blk, 0, stream>>>(W_in, WinT, 64, D_IN_PROJ, 64, D_PROJ_PAD);
    transpose_pad_bf16<<<(64 * 128 + 255) / 256, blk, 0, stream>>>(W_out, WoutT, D_INNER, 64, D_INNER, 64);

    // 1) h = xpad @ W_lin + b_lin       [32768 x 64], K=64
    gemm_ws_wmma<64, true><<<dim3(1, NTOK / 64), blk, 0, stream>>>(xpad, WlinT, b_lin, h, D_MODEL);

    // 2) zxbcdt = h @ W_in              [32768 x 384(pad)], K=64
    gemm_ws_wmma<64, false><<<dim3(D_PROJ_PAD / 64, NTOK / 64), blk, 0, stream>>>(h, WinT, nullptr, zxbcdt, D_PROJ_PAD);

    // 3) causal depthwise conv + silu -> xbc [32768 x 192]
    conv_silu_kernel<<<(NTOK * D_XBC + 255) / 256, blk, 0, stream>>>(zxbcdt, conv_w, conv_b, xbc);

    // 4) dt softplus + per-step decay
    dt_kernel<<<(BATCH * NHEADS * SEQ + 255) / 256, blk, 0, stream>>>(zxbcdt, dt_bias, A_log, dtv, decay);

    // 5) recurrent scan -> yraw (includes + D*x)
    scan_kernel<<<BATCH * NHEADS, 1024, 0, stream>>>(xbc, dtv, decay, Dp, yraw);

    // 6) gate with silu(z) + RMS norm (in place)
    gate_norm_kernel<<<NTOK / 8, blk, 0, stream>>>(yraw, zxbcdt, norm_w);

    // 7) yout = ynorm @ W_out           [32768 x 64], K=128
    gemm_ws_wmma<128, false><<<dim3(1, NTOK / 64), blk, 0, stream>>>(yraw, WoutT, nullptr, yout, D_MODEL);

    // 8) mean pool over L + classifier -> out [16 x 6]
    pool_cls_kernel<<<BATCH, blk, 0, stream>>>(yout, W_cls, b_cls, out);
}
